// DropBlock_65103114272821
// MI455X (gfx1250) — compile-verified, matrix operand order checked by reading, not run on hardware
//
#include <hip/hip_runtime.h>
#include <hip/hip_bf16.h>

typedef __attribute__((ext_vector_type(16))) _Float16 v16h;
typedef __attribute__((ext_vector_type(8)))  float    v8f;
typedef __attribute__((ext_vector_type(4)))  float    v4f;
typedef __attribute__((ext_vector_type(4)))  int      v4i;

#define HH 80
#define WW 80
#define PLANE (HH * WW)          // 6400 elements per (n,c) plane
#define QUADS (PLANE / 4)        // 1600 float4 per plane
#define BS 5                     // block_size
#define NT 320                   // threads per block (10 wave32)
#define VITERS (QUADS / NT)      // 5 float4 per thread (vectorized phases)
#define SEG 20                   // elements per thread in row/col mapping
#define SMW (WW + 4)             // 84: mask plane row stride (4 zero cols left)
#define HXW 81                   // hx row stride (odd -> conflict-free columns)
#define HXH (HH + 4)             // 84: hx rows (4 zero rows on top)
#define COUNT_M 26214400.0f     // 64*64*80*80

// 5-wide running max over a register window, branch-free.
__device__ __forceinline__ float max5(const float* w) {
    return fmaxf(fmaxf(fmaxf(w[0], w[1]), fmaxf(w[2], w[3])), w[4]);
}

__global__ void db_zero_counter(unsigned int* __restrict__ counter) {
    if (threadIdx.x == 0) *counter = 0u;
}

// ---------------------------------------------------------------------------
// Shared per-plane block-mask pipeline: load mask into left-padded LDS plane,
// horizontal 5-max (row mapping, register rolling window) into top-padded hx,
// vertical 5-max (column mapping, register rolling window).
// ---------------------------------------------------------------------------
struct DbSmem {
    float    sm[HH * SMW];   // padded mask plane; also reused as compact inv
    float    hx[HXH * HXW];  // padded horizontal max
};

__device__ __forceinline__ void db_load_and_hmax(
    DbSmem& s, const v4i* __restrict__ mask4, long base4, int tid) {
    // Zero left pad of sm (80 rows x 4 cols = 320 = NT entries).
    s.sm[(tid >> 2) * SMW + (tid & 3)] = 0.f;
    // Zero top pad of hx (4 rows x 81 = 324 entries).
    for (int z = tid; z < 4 * HXW; z += NT) s.hx[z] = 0.f;
    // Vectorized B128 load of the mask plane into padded layout: quad q holds
    // elements of row q/20, cols (q%20)*4 .. +3 (rows are multiples of 4 wide).
#pragma unroll
    for (int it = 0; it < VITERS; ++it) {
        int q = tid + it * NT;
        v4i mv = mask4[base4 + q];
        int i = q / SEG, j = (q % SEG) * 4;
        *(v4f*)&s.sm[i * SMW + 4 + j] = __builtin_convertvector(mv, v4f);
    }
    __syncthreads();

    // Horizontal pass, row mapping: thread owns row i, cols [j0, j0+20).
    const int i  = tid >> 2;
    const int j0 = (tid & 3) * SEG;
    float w[SEG + 4];
#pragma unroll
    for (int k = 0; k < SEG + 4; ++k) w[k] = s.sm[i * SMW + j0 + k];
#pragma unroll
    for (int k = 0; k < SEG; ++k)
        s.hx[(i + 4) * HXW + j0 + k] = max5(&w[k]);
    __syncthreads();
}

// ---------------------------------------------------------------------------
// Pass 1: count ones of the inverted block mask. Per-thread counts (<=20,
// exact in f16) are reduced with one v_wmma_f32_16x16x32_f16; the exact
// integer per-block partial is atomicAdd'ed into the global counter.
// ---------------------------------------------------------------------------
__global__ __launch_bounds__(NT) void db_count_kernel(
    const int* __restrict__ mask, unsigned int* __restrict__ counter) {
    __shared__ DbSmem s;
    __shared__ _Float16 scnt[NT];

    const int  tid   = threadIdx.x;
    const long base4 = (long)blockIdx.x * QUADS;
    db_load_and_hmax(s, (const v4i*)mask, base4, tid);

    // Vertical pass, column mapping: thread owns column j, rows [r0, r0+20).
    const int j  = tid % WW;
    const int r0 = (tid / WW) * SEG;
    float v[SEG + 4];
#pragma unroll
    for (int k = 0; k < SEG + 4; ++k) v[k] = s.hx[(r0 + k) * HXW + j];
    int cnt = 0;
#pragma unroll
    for (int k = 0; k < SEG; ++k) cnt += (max5(&v[k]) == 0.f) ? 1 : 0;

    scnt[tid] = (_Float16)cnt;       // 0..20, exact in f16
    __syncthreads();

    // Wave 0 (EXEC all-ones in-wave) reduces all 320 counts with one WMMA:
    // D = A(16x32) x B(32x16 ones) -> D[m,n] = rowsum(m); column n=0 lives in
    // lanes 0 (m=0..7) and 16 (m=8..15).
    if (tid < 32) {
        v16h a, bb;
#pragma unroll
        for (int k = 0; k < 16; ++k) bb[k] = (_Float16)1.0f;
#pragma unroll
        for (int k = 0; k < 10; ++k) a[k] = scnt[tid * 10 + k];
#pragma unroll
        for (int k = 10; k < 16; ++k) a[k] = (_Float16)0.0f;
        v8f c = {};
        c = __builtin_amdgcn_wmma_f32_16x16x32_f16(
            false, a, false, bb, (short)0, c, false, false);
        if ((tid & 15) == 0) {
            float sum = c[0] + c[1] + c[2] + c[3] + c[4] + c[5] + c[6] + c[7];
            atomicAdd(counter, (unsigned int)(sum + 0.5f));  // exact integer
        }
    }
}

// ---------------------------------------------------------------------------
// Pass 2: recompute the block mask (mask reads likely hit the 192 MB L2 after
// pass 1), write the 0/1 inverse mask into compact LDS layout, then stream
// out4 = inv4 * x4 * scale with B128 non-temporal load/store (x/out working
// set >> L2, so bypass).
// ---------------------------------------------------------------------------
__global__ __launch_bounds__(NT) void db_apply_kernel(
    const float* __restrict__ x, const int* __restrict__ mask,
    const unsigned int* __restrict__ counter, float* __restrict__ out) {
    __shared__ DbSmem s;

    const int  tid   = threadIdx.x;
    const long base4 = (long)blockIdx.x * QUADS;
    const float scale = COUNT_M / (float)(*counter);  // uniform scalar load
    db_load_and_hmax(s, (const v4i*)mask, base4, tid);

    // Vertical pass, column mapping; write inv into compact layout in s.sm
    // (padded sm is dead after the barrier inside db_load_and_hmax).
    const int j  = tid % WW;
    const int r0 = (tid / WW) * SEG;
    float v[SEG + 4];
#pragma unroll
    for (int k = 0; k < SEG + 4; ++k) v[k] = s.hx[(r0 + k) * HXW + j];
    float* inv_c = s.sm;             // compact [80*80] view
#pragma unroll
    for (int k = 0; k < SEG; ++k)
        inv_c[(r0 + k) * WW + j] = (max5(&v[k]) == 0.f) ? 1.0f : 0.0f;
    __syncthreads();

    // Streaming phase: fully vectorized, coalesced, L2-bypassing.
    const v4f* __restrict__ x4   = (const v4f*)x;
    v4f*       __restrict__ out4 = (v4f*)out;
    const v4f* inv4 = (const v4f*)inv_c;
#pragma unroll
    for (int it = 0; it < VITERS; ++it) {
        int q = tid + it * NT;
        v4f xv = __builtin_nontemporal_load(&x4[base4 + q]);
        v4f r  = inv4[q] * xv * scale;
        __builtin_nontemporal_store(r, &out4[base4 + q]);
    }
}

extern "C" void kernel_launch(void* const* d_in, const int* in_sizes, int n_in,
                              void* d_out, int out_size, void* d_ws, size_t ws_size,
                              hipStream_t stream) {
    const float* x    = (const float*)d_in[0];
    const int*   mask = (const int*)d_in[1];
    float*       out  = (float*)d_out;
    unsigned int* counter = (unsigned int*)d_ws;

    const int planes = in_sizes[0] / PLANE;  // 64*64 = 4096

    db_zero_counter<<<1, 64, 0, stream>>>(counter);
    db_count_kernel<<<planes, NT, 0, stream>>>(mask, counter);
    db_apply_kernel<<<planes, NT, 0, stream>>>(x, mask, counter, out);
}
